// PointNetPPEncoderFP_52819507806601
// MI455X (gfx1250) — matrix operand order repaired and checked by reading, not compiled, not run on hardware
//
#include <hip/hip_runtime.h>
#include <hip/hip_bf16.h>

// ---------------------------------------------------------------------------
// PointNet++ encoder forward, MI455X (gfx1250, wave32, WMMA).
//
// Pipeline:
//   1) pe_kernel       : positional encoding [B,N,9] (f32)
//   2) prep_weights x10: fold eval-BN into W/b, convert W -> bf16, pad K to 32
//   3) fps_kernel x2   : farthest point sampling (512 of 16384, 128 of 512)
//   4) sa1/sa2/fp2/fp1 : fused MLP stages; bf16 WMMA 16x16x32, f32 accum,
//                        point-major LDS activations, layers fused in LDS.
// ---------------------------------------------------------------------------

typedef __bf16 v16bf __attribute__((ext_vector_type(16)));
typedef float  v8f   __attribute__((ext_vector_type(8)));

struct __align__(16) U4   { unsigned int x, y, z, w; };
struct __align__(32) U4x2 { U4 lo, hi; };
struct __align__(16) F4   { float x, y, z, w; };

__device__ __forceinline__ unsigned short f2bf(float f) {
  unsigned int u = __builtin_bit_cast(unsigned int, f);
  u += 0x7FFFu + ((u >> 16) & 1u);            // round-to-nearest-even
  return (unsigned short)(u >> 16);
}

__device__ __forceinline__ v16bf ld16bf(const unsigned short* p0,
                                        const unsigned short* p1) {
  U4x2 t;
  t.lo = *(const U4*)p0;
  t.hi = *(const U4*)p1;
  return __builtin_bit_cast(v16bf, t);
}

// ---------------------------------------------------------------------------
// Wave-level 16x16 output tile, K accumulated in steps of 32 (bf16 WMMA).
// A (weights) from global: lane m=L%16, per ISA 16-bit A layout:
//   elems 0..7  = W[m][kb + 8*(L/16) + 0..7]
//   elems 8..15 = W[m][kb + 16 + 8*(L/16) + 0..7]
// B (activations) from point-major LDS: lane n=L%16:
//   elems 0..7  = act[n][kb + 8*(L/16) + 0..7]
//   elems 8..15 = act[n][kb + 16 + 8*(L/16) + 0..7]
// ---------------------------------------------------------------------------
template <int COUT, int KPAD, int INS>
__device__ __forceinline__ void gemm_tile(const unsigned short* __restrict__ Wp,
                                          const unsigned short* actIn,
                                          int tm, int tn, int lane, v8f& acc) {
  const int n  = lane & 15;
  const int kh = lane >> 4;
  const unsigned short* wrow = Wp + (size_t)(tm * 16 + n) * KPAD + kh * 8;
  const unsigned short* brow = actIn + (size_t)(tn * 16 + n) * INS + kh * 8;
#pragma unroll
  for (int kb = 0; kb < KPAD; kb += 32) {
    v16bf a = ld16bf(wrow + kb, wrow + kb + 16);
    v16bf b = ld16bf(brow + kb, brow + kb + 16);
    acc = __builtin_amdgcn_wmma_f32_16x16x32_bf16(false, a, false, b,
                                                  (short)0, acc, false, false);
  }
}

// bias + ReLU + pack to bf16, one b128 store (LDS or global, point-major).
template <int COUT, int KPAD, int INS>
__device__ __forceinline__ void layer_bf16(const unsigned short* __restrict__ Wp,
                                           const float* __restrict__ bias,
                                           const unsigned short* actIn,
                                           unsigned short* actOut, int outs) {
  const int lane = threadIdx.x & 31;
  const int wave = threadIdx.x >> 5;
  constexpr int NT = (COUT / 16) * 4;
  for (int t = wave; t < NT; t += 8) {
    const int tm = t >> 2, tn = t & 3;
    v8f acc = {};
    gemm_tile<COUT, KPAD, INS>(Wp, actIn, tm, tn, lane, acc);
    const int n = lane & 15, kh = lane >> 4;
    const float* bp = bias + tm * 16 + kh * 8;
    unsigned int pk[4];
#pragma unroll
    for (int j = 0; j < 4; ++j) {
      float x0 = fmaxf(acc[2 * j]     + bp[2 * j],     0.f);
      float x1 = fmaxf(acc[2 * j + 1] + bp[2 * j + 1], 0.f);
      pk[j] = (unsigned int)f2bf(x0) | ((unsigned int)f2bf(x1) << 16);
    }
    U4 u{pk[0], pk[1], pk[2], pk[3]};
    *(U4*)(actOut + (size_t)(tn * 16 + n) * outs + tm * 16 + kh * 8) = u;
  }
}

// bias + ReLU, f32 global store (final layer), two b128 stores per lane.
template <int COUT, int KPAD, int INS>
__device__ __forceinline__ void layer_f32(const unsigned short* __restrict__ Wp,
                                          const float* __restrict__ bias,
                                          const unsigned short* actIn,
                                          float* gout, int gstride) {
  const int lane = threadIdx.x & 31;
  const int wave = threadIdx.x >> 5;
  constexpr int NT = (COUT / 16) * 4;
  for (int t = wave; t < NT; t += 8) {
    const int tm = t >> 2, tn = t & 3;
    v8f acc = {};
    gemm_tile<COUT, KPAD, INS>(Wp, actIn, tm, tn, lane, acc);
    const int n = lane & 15, kh = lane >> 4;
    const float* bp = bias + tm * 16 + kh * 8;
    float* o = gout + (size_t)(tn * 16 + n) * gstride + tm * 16 + kh * 8;
    F4 lo{fmaxf(acc[0] + bp[0], 0.f), fmaxf(acc[1] + bp[1], 0.f),
          fmaxf(acc[2] + bp[2], 0.f), fmaxf(acc[3] + bp[3], 0.f)};
    F4 hi{fmaxf(acc[4] + bp[4], 0.f), fmaxf(acc[5] + bp[5], 0.f),
          fmaxf(acc[6] + bp[6], 0.f), fmaxf(acc[7] + bp[7], 0.f)};
    *(F4*)o = lo;
    *(F4*)(o + 4) = hi;
  }
}

// ---------------------------------------------------------------------------
// 1) positional encoding
// ---------------------------------------------------------------------------
__global__ __launch_bounds__(256) void pe_kernel(const float* __restrict__ xyz,
                                                 const float* __restrict__ nrm,
                                                 float* __restrict__ pe,
                                                 int total) {
  int i = blockIdx.x * 256 + threadIdx.x;
  if (i >= total) return;
  float x = xyz[i * 3 + 0], y = xyz[i * 3 + 1], z = xyz[i * 3 + 2];
  float r = sqrtf(x * x + y * y + z * z) + 1e-8f;
  float th = atan2f(y, x);
  float c = fminf(1.f, fmaxf(-1.f, z / r));
  float ph = acosf(c);
  float nx = nrm[i * 3 + 0], ny = nrm[i * 3 + 1], nz = nrm[i * 3 + 2];
  float inl = 1.f / (sqrtf(nx * nx + ny * ny + nz * nz) + 1e-12f);
  float* o = pe + (size_t)i * 9;
  o[0] = x; o[1] = y; o[2] = z; o[3] = r; o[4] = th; o[5] = ph;
  o[6] = nx * inl; o[7] = ny * inl; o[8] = nz * inl;
}

// ---------------------------------------------------------------------------
// 2) fold BN into weights, convert to bf16, zero-pad K to KPAD
// ---------------------------------------------------------------------------
__global__ __launch_bounds__(256) void prep_weights(
    const float* __restrict__ W, const float* __restrict__ b,
    const float* __restrict__ g, const float* __restrict__ beta,
    const float* __restrict__ mu, const float* __restrict__ var,
    int cout, int cin, int kpad, unsigned short* __restrict__ Wp,
    float* __restrict__ bp) {
  int i = blockIdx.x * 256 + threadIdx.x;
  if (i >= cout * kpad) return;
  int o = i / kpad, k = i - o * kpad;
  float s = g[o] * rsqrtf(var[o] + 1e-5f);
  float w = (k < cin) ? W[o * cin + k] * s : 0.f;
  Wp[i] = f2bf(w);
  if (k == 0) bp[o] = (b[o] - mu[o]) * s + beta[o];
}

// ---------------------------------------------------------------------------
// 3) farthest point sampling: one workgroup per batch, coords in registers,
//    LDS argmax tree; matches lax.scan semantics (idx[0]=0, centroid recorded
//    before the distance update).
// ---------------------------------------------------------------------------
template <int THREADS, int PPT>
__global__ __launch_bounds__(THREADS) void fps_kernel(
    const float* __restrict__ xyz, int N, int npoint, int* __restrict__ idx_out,
    float* __restrict__ sel_xyz) {
  __shared__ float sv[THREADS];
  __shared__ int   si[THREADS];
  __shared__ float cx, cy, cz;
  __shared__ int   scur;
  __shared__ int   sidxbuf[512];
  const int b = blockIdx.x;
  const int t = threadIdx.x;
  const float* base = xyz + (size_t)b * N * 3;
  float px[PPT], py[PPT], pz[PPT], dm[PPT];
#pragma unroll
  for (int i = 0; i < PPT; ++i) {
    int p = t + i * THREADS;
    px[i] = base[p * 3 + 0];
    py[i] = base[p * 3 + 1];
    pz[i] = base[p * 3 + 2];
    dm[i] = 1e10f;
  }
  if (t == 0) { scur = 0; cx = base[0]; cy = base[1]; cz = base[2]; }
  __syncthreads();
  for (int it = 0; it < npoint; ++it) {
    if (t == 0) sidxbuf[it] = scur;
    float bx = cx, by = cy, bz = cz;
    float bestv = -1.f;
    int besti = 0;
#pragma unroll
    for (int i = 0; i < PPT; ++i) {
      float dx = px[i] - bx, dy = py[i] - by, dz = pz[i] - bz;
      float d = dx * dx + dy * dy + dz * dz;
      dm[i] = fminf(dm[i], d);
      if (dm[i] > bestv) { bestv = dm[i]; besti = t + i * THREADS; }
    }
    sv[t] = bestv;
    si[t] = besti;
    __syncthreads();
    for (int s = THREADS / 2; s > 0; s >>= 1) {
      if (t < s && sv[t + s] > sv[t]) { sv[t] = sv[t + s]; si[t] = si[t + s]; }
      __syncthreads();
    }
    if (t == 0) scur = si[0];
    __syncthreads();
    int cur = scur;
    if (t == (cur % THREADS)) {
      int i = cur / THREADS;
      cx = px[i]; cy = py[i]; cz = pz[i];
    }
    __syncthreads();
  }
  for (int j = t; j < npoint; j += THREADS) {
    int id = sidxbuf[j];
    idx_out[b * npoint + j] = id;
    if (sel_xyz) {
      sel_xyz[((size_t)b * npoint + j) * 3 + 0] = base[id * 3 + 0];
      sel_xyz[((size_t)b * npoint + j) * 3 + 1] = base[id * 3 + 1];
      sel_xyz[((size_t)b * npoint + j) * 3 + 2] = base[id * 3 + 2];
    }
  }
}

// ---------------------------------------------------------------------------
// 4) fused MLP stages. TILE_N = 64 points per workgroup, 8 waves, point-major
//    LDS ping-pong buffers (stride padded +8 bf16 vs K for bank spread).
// ---------------------------------------------------------------------------
constexpr int TN = 64;
constexpr int SA1_S0 = 40, SA1_S1 = 72;
constexpr int SA2_S = 136;
constexpr int FP2_S = 264;
constexpr int FP1_S0 = 296, FP1_S1 = 136;
constexpr int SA1_LDS = (TN * SA1_S0 + 2 * TN * SA1_S1) * 2;
constexpr int SA2_LDS = 2 * TN * SA2_S * 2;
constexpr int FP2_LDS = 2 * TN * FP2_S * 2;
constexpr int FP1_LDS = (TN * FP1_S0 + TN * FP1_S1) * 2;

// sa1: gather pe at fps_idx1, 9->64->64->128, out bf16 [B,512,128]
__global__ __launch_bounds__(256) void sa1_kernel(
    const float* __restrict__ pe, const int* __restrict__ idx1,
    const unsigned short* W0, const float* B0, const unsigned short* W1,
    const float* B1, const unsigned short* W2, const float* B2,
    unsigned short* __restrict__ l1f) {
  extern __shared__ char smem[];
  unsigned short* bufIn = (unsigned short*)smem;
  unsigned short* bufA = bufIn + TN * SA1_S0;
  unsigned short* bufB = bufA + TN * SA1_S1;
  const int b = blockIdx.x >> 3;
  const int base = (blockIdx.x & 7) * TN;
  const int t = threadIdx.x;
  if (t == 0) { __builtin_prefetch(W0, 0, 0); __builtin_prefetch(W1, 0, 0); }
  if (t < TN) {
    int src = idx1[b * 512 + base + t];
    const float* row = pe + ((size_t)b * 16384 + src) * 9;
    unsigned short* d = bufIn + t * SA1_S0;
#pragma unroll
    for (int c = 0; c < 9; ++c) d[c] = f2bf(row[c]);
#pragma unroll
    for (int c = 9; c < 32; ++c) d[c] = 0;
  }
  __syncthreads();
  layer_bf16<64, 32, SA1_S0>(W0, B0, bufIn, bufA, SA1_S1);
  __syncthreads();
  layer_bf16<64, 64, SA1_S1>(W1, B1, bufA, bufB, SA1_S1);
  __syncthreads();
  layer_bf16<128, 64, SA1_S1>(W2, B2, bufB,
                              l1f + ((size_t)b * 512 + base) * 128, 128);
}

// sa2: gather l1f at fps_idx2, 128->128->128->128, out bf16 [B,128,128]
__global__ __launch_bounds__(256) void sa2_kernel(
    const unsigned short* __restrict__ l1f, const int* __restrict__ idx2,
    const unsigned short* W0, const float* B0, const unsigned short* W1,
    const float* B1, const unsigned short* W2, const float* B2,
    unsigned short* __restrict__ l2f) {
  extern __shared__ char smem[];
  unsigned short* X = (unsigned short*)smem;
  unsigned short* Y = X + TN * SA2_S;
  const int b = blockIdx.x >> 1;
  const int base = (blockIdx.x & 1) * TN;
  if (threadIdx.x == 0) __builtin_prefetch(W0, 0, 0);
  for (int c = threadIdx.x; c < TN * 16; c += 256) {
    int pt = c >> 4, j = c & 15;
    int src = idx2[b * 128 + base + pt];
    *(U4*)(X + pt * SA2_S + j * 8) =
        *(const U4*)(l1f + ((size_t)b * 512 + src) * 128 + j * 8);
  }
  __syncthreads();
  layer_bf16<128, 128, SA2_S>(W0, B0, X, Y, SA2_S);
  __syncthreads();
  layer_bf16<128, 128, SA2_S>(W1, B1, Y, X, SA2_S);
  __syncthreads();
  layer_bf16<128, 128, SA2_S>(W2, B2, X,
                              l2f + ((size_t)b * 128 + base) * 128, 128);
}

// fp2: concat [l1f ; upsample4(l2f)] (256ch) -> 256 -> 256, out bf16 [B,512,256]
__global__ __launch_bounds__(256) void fp2_kernel(
    const unsigned short* __restrict__ l1f,
    const unsigned short* __restrict__ l2f, const unsigned short* W0,
    const float* B0, const unsigned short* W1, const float* B1,
    unsigned short* __restrict__ l1fp) {
  extern __shared__ char smem[];
  unsigned short* X = (unsigned short*)smem;
  unsigned short* Y = X + TN * FP2_S;
  const int b = blockIdx.x >> 3;
  const int base = (blockIdx.x & 7) * TN;
  if (threadIdx.x == 0) __builtin_prefetch(W0, 0, 0);
  for (int c = threadIdx.x; c < TN * 32; c += 256) {
    int pt = c >> 5, j = c & 31;
    int p = base + pt;
    const unsigned short* src =
        (j < 16) ? l1f + ((size_t)b * 512 + p) * 128 + j * 8
                 : l2f + ((size_t)b * 128 + (p >> 2)) * 128 + (j - 16) * 8;
    *(U4*)(X + pt * FP2_S + j * 8) = *(const U4*)src;
  }
  __syncthreads();
  layer_bf16<256, 256, FP2_S>(W0, B0, X, Y, FP2_S);
  __syncthreads();
  layer_bf16<256, 256, FP2_S>(W1, B1, Y,
                              l1fp + ((size_t)b * 512 + base) * 256, 256);
}

// fp1: concat [upsample32(l1fp) ; pe] (265ch pad 288) -> 128 -> 128, f32 out
__global__ __launch_bounds__(256) void fp1_kernel(
    const unsigned short* __restrict__ l1fp, const float* __restrict__ pe,
    const unsigned short* W0, const float* B0, const unsigned short* W1,
    const float* B1, float* __restrict__ out) {
  extern __shared__ char smem[];
  unsigned short* X = (unsigned short*)smem;
  unsigned short* Y = X + TN * FP1_S0;
  const int b = blockIdx.x >> 8;
  const int base = (blockIdx.x & 255) * TN;
  if (threadIdx.x == 0) __builtin_prefetch(W0, 0, 0);
  for (int c = threadIdx.x; c < TN * 32; c += 256) {
    int pt = c >> 5, j = c & 31;
    int p = base + pt;
    *(U4*)(X + pt * FP1_S0 + j * 8) =
        *(const U4*)(l1fp + ((size_t)b * 512 + (p >> 5)) * 256 + j * 8);
  }
  if (threadIdx.x < TN) {
    int p = base + threadIdx.x;
    const float* row = pe + ((size_t)b * 16384 + p) * 9;
    unsigned short* d = X + threadIdx.x * FP1_S0 + 256;
#pragma unroll
    for (int c = 0; c < 9; ++c) d[c] = f2bf(row[c]);
#pragma unroll
    for (int c = 9; c < 32; ++c) d[c] = 0;
  }
  __syncthreads();
  layer_bf16<128, 288, FP1_S0>(W0, B0, X, Y, FP1_S1);
  __syncthreads();
  layer_f32<128, 128, FP1_S1>(W1, B1, Y,
                              out + ((size_t)b * 16384 + base) * 128, 128);
}

// ---------------------------------------------------------------------------
// host launcher
// ---------------------------------------------------------------------------
extern "C" void kernel_launch(void* const* d_in, const int* in_sizes, int n_in,
                              void* d_out, int out_size, void* d_ws,
                              size_t ws_size, hipStream_t stream) {
  (void)in_sizes; (void)n_in; (void)out_size; (void)ws_size;
  constexpr int B = 16, N = 16384, NP1 = 512, NP2 = 128;
  const float* xyz = (const float*)d_in[0];
  const float* nrm = (const float*)d_in[1];

  struct LD { int cout, cin, kpad; };
  static const LD L[10] = {
      {64, 9, 32},    {64, 64, 64},   {128, 64, 64},   // sa1
      {128, 128, 128},{128, 128, 128},{128, 128, 128}, // sa2
      {256, 256, 256},{256, 256, 256},                 // fp2
      {128, 265, 288},{128, 128, 128},                 // fp1
  };

  char* ws = (char*)d_ws;
  size_t off = 0;
  auto alloc = [&](size_t bytes) -> void* {
    void* p = ws + off;
    off = (off + bytes + 255) & ~(size_t)255;
    return p;
  };

  float* pe = (float*)alloc((size_t)B * N * 9 * 4);
  unsigned short* Wp[10];
  float* Bp[10];
  for (int i = 0; i < 10; ++i) {
    Wp[i] = (unsigned short*)alloc((size_t)L[i].cout * L[i].kpad * 2);
    Bp[i] = (float*)alloc((size_t)L[i].cout * 4);
  }
  int* idx1 = (int*)alloc((size_t)B * NP1 * 4);
  float* sxyz = (float*)alloc((size_t)B * NP1 * 3 * 4);
  int* idx2 = (int*)alloc((size_t)B * NP2 * 4);
  unsigned short* l1f = (unsigned short*)alloc((size_t)B * NP1 * 128 * 2);
  unsigned short* l2f = (unsigned short*)alloc((size_t)B * NP2 * 128 * 2);
  unsigned short* l1fp = (unsigned short*)alloc((size_t)B * NP1 * 256 * 2);

  pe_kernel<<<(B * N + 255) / 256, 256, 0, stream>>>(xyz, nrm, pe, B * N);

  for (int i = 0; i < 10; ++i) {
    const float* w    = (const float*)d_in[2 + i * 6 + 0];
    const float* bb   = (const float*)d_in[2 + i * 6 + 1];
    const float* g    = (const float*)d_in[2 + i * 6 + 2];
    const float* beta = (const float*)d_in[2 + i * 6 + 3];
    const float* mu   = (const float*)d_in[2 + i * 6 + 4];
    const float* var  = (const float*)d_in[2 + i * 6 + 5];
    int tot = L[i].cout * L[i].kpad;
    prep_weights<<<(tot + 255) / 256, 256, 0, stream>>>(
        w, bb, g, beta, mu, var, L[i].cout, L[i].cin, L[i].kpad, Wp[i], Bp[i]);
  }

  fps_kernel<1024, 16><<<B, 1024, 0, stream>>>(xyz, N, NP1, idx1, sxyz);
  fps_kernel<256, 2><<<B, 256, 0, stream>>>(sxyz, NP1, NP2, idx2, nullptr);

  sa1_kernel<<<B * 8, 256, SA1_LDS, stream>>>(pe, idx1, Wp[0], Bp[0], Wp[1],
                                              Bp[1], Wp[2], Bp[2], l1f);
  sa2_kernel<<<B * 2, 256, SA2_LDS, stream>>>(l1f, idx2, Wp[3], Bp[3], Wp[4],
                                              Bp[4], Wp[5], Bp[5], l2f);
  fp2_kernel<<<B * 8, 256, FP2_LDS, stream>>>(l1f, l2f, Wp[6], Bp[6], Wp[7],
                                              Bp[7], l1fp);
  fp1_kernel<<<B * 256, 256, FP1_LDS, stream>>>(l1fp, pe, Wp[8], Bp[8], Wp[9],
                                                Bp[9], (float*)d_out);
}
